// TransformerBlock_9526237462713
// MI455X (gfx1250) — matrix operand order repaired and checked by reading, not compiled, not run on hardware
//
#include <hip/hip_runtime.h>

// ---------------------------------------------------------------------------
// Transformer block for MI455X (gfx1250), bf16 WMMA (16x16x32) with f32 accum.
// GEMMs stage B tiles through LDS via the Tensor Data Mover (TDM),
// double-buffered on TENSORcnt. B=4, C=2048, D=768, H=12, HD=64
// ---------------------------------------------------------------------------

#define Bq  4
#define Cq  2048
#define Dq  768
#define Hq  12
#define HDq 64

typedef __bf16 bf16;
typedef unsigned int u32;
typedef unsigned long long u64;
typedef __attribute__((ext_vector_type(16))) __bf16 v16bf;
typedef __attribute__((ext_vector_type(8)))  __bf16 v8bf;
typedef __attribute__((ext_vector_type(4)))  __bf16 v4bf;
typedef __attribute__((ext_vector_type(8)))  float  v8f;
typedef __attribute__((ext_vector_type(4)))  u32    v4u;
typedef __attribute__((ext_vector_type(8)))  int    v8i;
typedef __attribute__((ext_vector_type(4)))  int    v4i;

// Padded LDS row stride for TDM-staged B tiles: 32 elems + 8 pad elems (16B pad
// per 64B interval) -> 80B rows for bank spread.
#define BROW 40

__device__ __forceinline__ bf16 f2bf(float f) {
    unsigned u = __float_as_uint(f);
    unsigned r = (u + 0x7fffu + ((u >> 16) & 1u)) >> 16;   // round-to-nearest-even
    unsigned short s = (unsigned short)r;
    return __builtin_bit_cast(bf16, s);
}

// A fragment (16x32, row-major source, contiguous k order):
// lanes 0-15: row=lane,    K = 0..15 ; lanes16-31: K = 16..31
__device__ __forceinline__ v16bf load_a_frag(const bf16* A, int lda, int rowBase,
                                             int kBase, int lane) {
    int r = lane & 15, kh = lane >> 4;
    const bf16* p0 = A + (size_t)(rowBase + r) * lda + kBase + kh * 16;
    v8bf x0 = *(const v8bf*)p0;
    v8bf x1 = *(const v8bf*)(p0 + 8);
    v16bf a;
#pragma unroll
    for (int i = 0; i < 8; ++i) { a[i] = x0[i]; a[8 + i] = x1[i]; }
    return a;
}

// B fragment (32x16) from transposed weights Bt (N x K row-major), global memory.
__device__ __forceinline__ v16bf load_b_frag(const bf16* Bt, int ldb, int nBase,
                                             int kBase, int lane) {
    int n = lane & 15, kh = lane >> 4;
    const bf16* p = Bt + (size_t)(nBase + n) * ldb + kBase + kh * 16;
    v8bf x0 = *(const v8bf*)p;
    v8bf x1 = *(const v8bf*)(p + 8);
    v16bf b;
#pragma unroll
    for (int i = 0; i < 8; ++i) { b[i] = x0[i]; b[8 + i] = x1[i]; }
    return b;
}

// B fragment from a TDM-staged LDS tile (row stride BROW elems, 8B-aligned rows).
__device__ __forceinline__ v16bf load_b_frag_lds(const bf16* tile, int nSub, int lane) {
    int n = lane & 15, kh = lane >> 4;
    const bf16* p = tile + (nSub * 16 + n) * BROW + kh * 16;
    v4bf x0 = *(const v4bf*)p;
    v4bf x1 = *(const v4bf*)(p + 4);
    v4bf x2 = *(const v4bf*)(p + 8);
    v4bf x3 = *(const v4bf*)(p + 12);
    v16bf b;
#pragma unroll
    for (int i = 0; i < 4; ++i) {
        b[i] = x0[i]; b[4 + i] = x1[i]; b[8 + i] = x2[i]; b[12 + i] = x3[i];
    }
    return b;
}

#define WMMA_BF16(A, Bf, C) \
    __builtin_amdgcn_wmma_f32_16x16x32_bf16(false, (A), false, (Bf), (short)0, (C), false, false)

// ---------------------------------------------------------------------------
// TDM: DMA a 64-row x 32-elem bf16 tile (row stride strideElems) from global
// memory into LDS at lds_off, inserting 16B padding after every 64B row
// (D# per CDNA5 ISA). Issued once per wave.
// 6-arg builtin form (amdgpu-toolchain clang-23).
// ---------------------------------------------------------------------------
__device__ __forceinline__ void tdm_load_tile(u32 lds_off, const bf16* gptr,
                                              int strideElems) {
    u64 ga   = (u64)(uintptr_t)gptr;
    u64 g0q0 = 1ull | ((u64)lds_off << 32);                    // count=1 | lds_addr
    u64 g0q1 = (ga & ((1ull << 57) - 1)) | (2ull << 62);       // global_addr | type=2
    v4u g0;
    g0[0] = (u32)g0q0; g0[1] = (u32)(g0q0 >> 32);
    g0[2] = (u32)g0q1; g0[3] = (u32)(g0q1 >> 32);

    u64 td0 = 32, td1 = 64, tile0 = 32, tile1 = 64;
    u64 st0 = (u64)(u32)strideElems;
    // q0: wg_mask=0 | data_size=1(2B)<<16 | pad_en<<20 | pad_int=3(16dw)<<22 |
    //     pad_amt=3(4dw)<<25 | abar_addr=0 | tensor_dim0[15:0]<<48
    u64 q0 = (1ull << 16) | (1ull << 20) | (3ull << 22) | (3ull << 25)
           | ((td0 & 0xFFFF) << 48);
    u64 q1 = ((td0 >> 16) & 0xFFFF) | (td1 << 16) | (tile0 << 48);
    u64 q2 = tile1 | ((st0 & 0xFFFFFFFFull) << 32);            // tile_dim1 | stride0.lo
    u64 q3 = (st0 >> 32) & 0xFFFF;                             // stride0.hi | stride1=0
    v8i g1;
    g1[0] = (int)(u32)q0; g1[1] = (int)(u32)(q0 >> 32);
    g1[2] = (int)(u32)q1; g1[3] = (int)(u32)(q1 >> 32);
    g1[4] = (int)(u32)q2; g1[5] = (int)(u32)(q2 >> 32);
    g1[6] = (int)(u32)q3; g1[7] = (int)(u32)(q3 >> 32);

    v4i g2 = {};                                               // dims 2+ unused
    v4i g3 = {};
    v8i g4 = {};                                               // unused trailing group
    __builtin_amdgcn_tensor_load_to_lds(g0, g1, g2, g3, g4, 0);
}

// ---------------------------------------------------------------------------
// Weight transpose + f32->bf16:  out[n*K + k] = in[k*N + n]
// ---------------------------------------------------------------------------
__global__ void transpose_bf16_kernel(const float* __restrict__ in,
                                      bf16* __restrict__ out, int K, int N) {
    int idx = blockIdx.x * blockDim.x + threadIdx.x;
    if (idx >= K * N) return;
    int k = idx / N, n = idx % N;
    out[(size_t)n * K + k] = f2bf(in[idx]);
}

// ---------------------------------------------------------------------------
// LayerNorm over D=768, one block per row, 256 threads (8 waves).
// ---------------------------------------------------------------------------
__global__ void ln_kernel(const float* __restrict__ x, const float* __restrict__ g,
                          const float* __restrict__ bb, float* __restrict__ yf,
                          bf16* __restrict__ yb) {
    int row = blockIdx.x;
    const float* xr = x + (size_t)row * Dq;
    int tid = threadIdx.x, wave = tid >> 5, lane = tid & 31;
    float v[3], s = 0.f, s2 = 0.f;
#pragma unroll
    for (int i = 0; i < 3; ++i) { v[i] = xr[tid + i * 256]; s += v[i]; s2 += v[i] * v[i]; }
#pragma unroll
    for (int m = 16; m >= 1; m >>= 1) { s += __shfl_xor(s, m); s2 += __shfl_xor(s2, m); }
    __shared__ float sh[16];
    if (lane == 0) { sh[wave] = s; sh[8 + wave] = s2; }
    __syncthreads();
    s = 0.f; s2 = 0.f;
#pragma unroll
    for (int i = 0; i < 8; ++i) { s += sh[i]; s2 += sh[8 + i]; }
    float mu  = s * (1.f / Dq);
    float var = s2 * (1.f / Dq) - mu * mu;
    float rs  = rsqrtf(var + 1e-5f);
#pragma unroll
    for (int i = 0; i < 3; ++i) {
        int e = tid + i * 256;
        float o = (v[i] - mu) * rs * g[e] + bb[e];
        yf[(size_t)row * Dq + e] = o;
        yb[(size_t)row * Dq + e] = f2bf(o);
    }
}

// ---------------------------------------------------------------------------
// QKV projection: per (b,h), X_h(16x64) @ w_qkv(64x192). V written transposed.
// grid = (C/128, H, B), 8 waves x 16 rows. The 12 output subtiles are split
// into three statically-partitioned loops (Q / K / V^T) so store targets are
// compile-time known (straight-line epilogue, no per-element branches).
// ---------------------------------------------------------------------------
__global__ void qkv_kernel(const bf16* __restrict__ x1b, const bf16* __restrict__ wqkvT,
                           const float* __restrict__ bqkv, bf16* __restrict__ Q,
                           bf16* __restrict__ Kmat, bf16* __restrict__ Vt) {
    int ctile = blockIdx.x, h = blockIdx.y, b = blockIdx.z;
    int tid = threadIdx.x, wave = tid >> 5, lane = tid & 31;
    int half = lane >> 4, n = lane & 15;
    int cBase   = ctile * 128 + wave * 16;
    int rowBase = b * Cq + cBase;
    v16bf a0 = load_a_frag(x1b + h * HDq, Dq, rowBase, 0,  lane);
    v16bf a1 = load_a_frag(x1b + h * HDq, Dq, rowBase, 32, lane);
    size_t bhOff = ((size_t)b * Hq + h) * Cq * HDq;

    // Q: output cols 0..63
#pragma unroll 1
    for (int j = 0; j < 4; ++j) {
        v8f acc = {};
        v16bf b0 = load_b_frag(wqkvT, HDq, j * 16, 0, lane);
        acc = WMMA_BF16(a0, b0, acc);
        v16bf b1 = load_b_frag(wqkvT, HDq, j * 16, 32, lane);
        acc = WMMA_BF16(a1, b1, acc);
        int e = j * 16 + n;
        float bias = bqkv[e];
#pragma unroll
        for (int v = 0; v < 8; ++v) {
            int c = cBase + v + half * 8;
            Q[bhOff + (size_t)c * HDq + e] = f2bf(acc[v] + bias);
        }
    }
    // K: output cols 64..127
#pragma unroll 1
    for (int j = 4; j < 8; ++j) {
        v8f acc = {};
        v16bf b0 = load_b_frag(wqkvT, HDq, j * 16, 0, lane);
        acc = WMMA_BF16(a0, b0, acc);
        v16bf b1 = load_b_frag(wqkvT, HDq, j * 16, 32, lane);
        acc = WMMA_BF16(a1, b1, acc);
        int e = j * 16 + n;
        float bias = bqkv[e];
        int ek = e - 64;
#pragma unroll
        for (int v = 0; v < 8; ++v) {
            int c = cBase + v + half * 8;
            Kmat[bhOff + (size_t)c * HDq + ek] = f2bf(acc[v] + bias);
        }
    }
    // V (transposed, HD x C): output cols 128..191
#pragma unroll 1
    for (int j = 8; j < 12; ++j) {
        v8f acc = {};
        v16bf b0 = load_b_frag(wqkvT, HDq, j * 16, 0, lane);
        acc = WMMA_BF16(a0, b0, acc);
        v16bf b1 = load_b_frag(wqkvT, HDq, j * 16, 32, lane);
        acc = WMMA_BF16(a1, b1, acc);
        int e = j * 16 + n;
        float bias = bqkv[e];
        int ev = e - 128;
#pragma unroll
        for (int v = 0; v < 8; ++v) {
            int c = cBase + v + half * 8;
            Vt[bhOff + (size_t)ev * Cq + c] = f2bf(acc[v] + bias);
        }
    }
}

// ---------------------------------------------------------------------------
// Flash attention with the reference's MULTIPLICATIVE strict-lower mask:
// masked scores become 0 (not -inf) and still participate in softmax and P@V,
// so we iterate ALL key tiles. grid = (C/128, H, B); wave = 16 queries.
// ---------------------------------------------------------------------------
__global__ void attn_kernel(const bf16* __restrict__ Q, const bf16* __restrict__ Kmat,
                            const bf16* __restrict__ Vt, bf16* __restrict__ O) {
    __shared__ float lds[8 * 16 * 32];               // per-wave 16x32 P scratch
    int ctile = blockIdx.x, h = blockIdx.y, b = blockIdx.z;
    int tid = threadIdx.x, wave = tid >> 5, lane = tid & 31;
    int half = lane >> 4, n = lane & 15;
    size_t bhOff = ((size_t)b * Hq + h) * Cq * HDq;
    const bf16* Qbh = Q + bhOff;
    const bf16* Kbh = Kmat + bhOff;
    const bf16* Vbh = Vt + bhOff;
    int qBase = ctile * 128 + wave * 16;
    v16bf aq0 = load_a_frag(Qbh, HDq, qBase, 0,  lane);
    v16bf aq1 = load_a_frag(Qbh, HDq, qBase, 32, lane);
    v8f accO[4] = {};
    float mrow[8], lrow[8];
#pragma unroll
    for (int v = 0; v < 8; ++v) { mrow[v] = -3.0e38f; lrow[v] = 0.f; }
    float* ldsW = lds + wave * 512;

    for (int kt = 0; kt < Cq / 32; ++kt) {
        int keyBase = kt * 32;
        v8f s0 = {}, s1 = {};
        {
            v16bf bk;
            bk = load_b_frag(Kbh, HDq, keyBase,      0,  lane); s0 = WMMA_BF16(aq0, bk, s0);
            bk = load_b_frag(Kbh, HDq, keyBase,      32, lane); s0 = WMMA_BF16(aq1, bk, s0);
            bk = load_b_frag(Kbh, HDq, keyBase + 16, 0,  lane); s1 = WMMA_BF16(aq0, bk, s1);
            bk = load_b_frag(Kbh, HDq, keyBase + 16, 32, lane); s1 = WMMA_BF16(aq1, bk, s1);
        }
#pragma unroll
        for (int v = 0; v < 8; ++v) {
            int rowLocal = v + half * 8;
            int row = qBase + rowLocal;
            float e0 = s0[v] * 0.125f, e1 = s1[v] * 0.125f;
            if (keyBase + n      >= row) e0 = 0.f;
            if (keyBase + 16 + n >= row) e1 = 0.f;
            float t = fmaxf(e0, e1);
            t = fmaxf(t, __shfl_xor(t, 1)); t = fmaxf(t, __shfl_xor(t, 2));
            t = fmaxf(t, __shfl_xor(t, 4)); t = fmaxf(t, __shfl_xor(t, 8));
            float mNew = fmaxf(mrow[v], t);
            float p0 = __expf(e0 - mNew), p1 = __expf(e1 - mNew);
            float rs = p0 + p1;
            rs += __shfl_xor(rs, 1); rs += __shfl_xor(rs, 2);
            rs += __shfl_xor(rs, 4); rs += __shfl_xor(rs, 8);
            float alpha = __expf(mrow[v] - mNew);
            lrow[v] = lrow[v] * alpha + rs;
            mrow[v] = mNew;
            ldsW[rowLocal * 32 + n]      = p0;
            ldsW[rowLocal * 32 + 16 + n] = p1;
#pragma unroll
            for (int nf = 0; nf < 4; ++nf) accO[nf][v] *= alpha;
        }
        // per-wave LDS transpose C-layout -> A-layout (DS ops in-order per wave)
        asm volatile("s_wait_dscnt 0" ::: "memory");
        v16bf ap;
        {
            int r = lane & 15, kh = lane >> 4;
            const float* pr = lds + wave * 512 + r * 32;
#pragma unroll
            for (int i = 0; i < 8; ++i) {
                ap[i]     = f2bf(pr[kh * 16 + i]);
                ap[8 + i] = f2bf(pr[kh * 16 + 8 + i]);
            }
        }
#pragma unroll
        for (int nf = 0; nf < 4; ++nf) {
            v16bf bv = load_b_frag(Vbh, Cq, nf * 16, keyBase, lane);
            accO[nf] = WMMA_BF16(ap, bv, accO[nf]);
        }
    }
#pragma unroll
    for (int nf = 0; nf < 4; ++nf)
#pragma unroll
        for (int v = 0; v < 8; ++v) {
            int c = qBase + v + half * 8;
            float val = accO[nf][v] / lrow[v];
            O[((size_t)b * Cq + c) * Dq + h * HDq + nf * 16 + n] = f2bf(val);
        }
}

// ---------------------------------------------------------------------------
// Generic GEMM with TDM-staged, double-buffered B tiles in LDS.
// out = epilogue(A(MxK,bf16) @ Bt(NxK,bf16)^T + bias [,gelu] [,+resid])
// grid = (M/128, N/64); 8 waves x 16 rows x 64 cols.
// ---------------------------------------------------------------------------
__global__ void gemm_kernel(const bf16* __restrict__ A, const bf16* __restrict__ Bt,
                            const float* __restrict__ bias, const float* __restrict__ resid,
                            float* __restrict__ outF, bf16* __restrict__ outB,
                            int Kdim, int N, int geluFlag) {
    __shared__ bf16 btile[2][64 * BROW];             // 2 x 5KB TDM-padded tiles
    int tid = threadIdx.x, wave = tid >> 5, lane = tid & 31;
    int half = lane >> 4, n = lane & 15;
    int rowBase = blockIdx.x * 128 + wave * 16;
    int colBase = blockIdx.y * 64;
    const bf16* Bblk = Bt + (size_t)colBase * Kdim;  // 64 rows of Bt
    int nSteps = Kdim / 32;

    if (wave == 0)
        tdm_load_tile((u32)(uintptr_t)(void*)&btile[0][0], Bblk, Kdim);

    v8f acc[4] = {};
#pragma unroll 1
    for (int ks = 0; ks < nSteps; ++ks) {
        if (ks + 1 < nSteps) {
            if (wave == 0) {
                tdm_load_tile((u32)(uintptr_t)(void*)&btile[(ks + 1) & 1][0],
                              Bblk + (ks + 1) * 32, Kdim);
                __builtin_amdgcn_s_wait_tensorcnt(1);   // tile ks complete
            }
        } else if (wave == 0) {
            __builtin_amdgcn_s_wait_tensorcnt(0);
        }
        __syncthreads();
        const bf16* cur = &btile[ks & 1][0];
        v16bf a = load_a_frag(A, Kdim, rowBase, ks * 32, lane);
#pragma unroll
        for (int j = 0; j < 4; ++j) {
            v16bf bfr = load_b_frag_lds(cur, j, lane);
            acc[j] = WMMA_BF16(a, bfr, acc[j]);
        }
        __syncthreads();                             // protect buffer for tile ks+2
    }
#pragma unroll
    for (int j = 0; j < 4; ++j) {
        int col = colBase + j * 16 + n;
        float bv = bias[col];
#pragma unroll
        for (int v = 0; v < 8; ++v) {
            int row = rowBase + v + half * 8;
            float val = acc[j][v] + bv;
            if (geluFlag) {
                float u = val, cg = 0.7978845608028654f;
                val = 0.5f * u * (1.f + tanhf(cg * (u + 0.044715f * u * u * u)));
            }
            size_t idx = (size_t)row * N + col;
            if (resid) val += resid[idx];
            if (outF) outF[idx] = val;
            if (outB) outB[idx] = f2bf(val);
        }
    }
}

// ---------------------------------------------------------------------------
extern "C" void kernel_launch(void* const* d_in, const int* in_sizes, int n_in,
                              void* d_out, int out_size, void* d_ws, size_t ws_size,
                              hipStream_t stream) {
    (void)in_sizes; (void)n_in; (void)out_size; (void)ws_size;
    const float* x     = (const float*)d_in[0];
    const float* ln1_g = (const float*)d_in[1];
    const float* ln1_b = (const float*)d_in[2];
    const float* ln2_g = (const float*)d_in[3];
    const float* ln2_b = (const float*)d_in[4];
    const float* w_qkv = (const float*)d_in[5];
    const float* b_qkv = (const float*)d_in[6];
    const float* w_ao  = (const float*)d_in[7];
    const float* b_ao  = (const float*)d_in[8];
    const float* w_ff1 = (const float*)d_in[9];
    const float* b_ff1 = (const float*)d_in[10];
    const float* w_ff2 = (const float*)d_in[11];
    const float* b_ff2 = (const float*)d_in[12];
    float* outp = (float*)d_out;

    char* p = (char*)d_ws;
    auto alloc = [&](size_t bytes) {
        char* r = p; p += (bytes + 255) & ~(size_t)255; return (void*)r;
    };
    const size_t BC = (size_t)Bq * Cq;                  // 8192
    float* x1    = (float*)alloc(BC * Dq * 4);
    bf16*  x1b   = (bf16*) alloc(BC * Dq * 2);
    float* x2    = (float*)alloc(BC * Dq * 4);
    float* x3    = (float*)alloc(BC * Dq * 4);
    bf16*  x3b   = (bf16*) alloc(BC * Dq * 2);
    bf16*  Qb    = (bf16*) alloc(BC * Dq * 2);          // (B,H,C,HD)
    bf16*  Kb    = (bf16*) alloc(BC * Dq * 2);          // (B,H,C,HD)
    bf16*  Vtb   = (bf16*) alloc(BC * Dq * 2);          // (B,H,HD,C)
    bf16*  aO    = (bf16*) alloc(BC * Dq * 2);          // attention out (B,C,D)
    bf16*  hb    = (bf16*) alloc(BC * 4 * Dq * 2);      // FF1 hidden
    bf16*  wqkvT = (bf16*) alloc((size_t)192 * 64 * 2);
    bf16*  waoT  = (bf16*) alloc((size_t)768 * 768 * 2);
    bf16*  wff1T = (bf16*) alloc((size_t)3072 * 768 * 2);
    bf16*  wff2T = (bf16*) alloc((size_t)768 * 3072 * 2);

    auto tg = [](int total) { return dim3((unsigned)((total + 255) / 256)); };
    transpose_bf16_kernel<<<tg(64 * 192),   256, 0, stream>>>(w_qkv, wqkvT, 64,   192);
    transpose_bf16_kernel<<<tg(768 * 768),  256, 0, stream>>>(w_ao,  waoT,  768,  768);
    transpose_bf16_kernel<<<tg(768 * 3072), 256, 0, stream>>>(w_ff1, wff1T, 768,  3072);
    transpose_bf16_kernel<<<tg(3072 * 768), 256, 0, stream>>>(w_ff2, wff2T, 3072, 768);

    ln_kernel<<<dim3((unsigned)BC), 256, 0, stream>>>(x, ln1_g, ln1_b, x1, x1b);
    qkv_kernel<<<dim3(Cq / 128, Hq, Bq), 256, 0, stream>>>(x1b, wqkvT, b_qkv, Qb, Kb, Vtb);
    attn_kernel<<<dim3(Cq / 128, Hq, Bq), 256, 0, stream>>>(Qb, Kb, Vtb, aO);
    // x2 = x1 + attn(x1) @ w_ao + b_ao
    gemm_kernel<<<dim3((unsigned)(BC / 128), Dq / 64), 256, 0, stream>>>(
        aO, waoT, b_ao, x1, x2, nullptr, Dq, Dq, 0);
    ln_kernel<<<dim3((unsigned)BC), 256, 0, stream>>>(x2, ln2_g, ln2_b, x3, x3b);
    // h = gelu(x3 @ w_ff1 + b_ff1)
    gemm_kernel<<<dim3((unsigned)(BC / 128), 4 * Dq / 64), 256, 0, stream>>>(
        x3b, wff1T, b_ff1, nullptr, nullptr, hb, Dq, 4 * Dq, 1);
    // out = x3 + h @ w_ff2 + b_ff2
    gemm_kernel<<<dim3((unsigned)(BC / 128), Dq / 64), 256, 0, stream>>>(
        hb, wff2T, b_ff2, x3, outp, nullptr, 4 * Dq, Dq, 0);
}
